// MvCnnLeftLayer_51797305590251
// MI455X (gfx1250) — compile-verified, matrix operand order checked by reading, not run on hardware
//
#include <hip/hip_runtime.h>
#include <stdint.h>

typedef __attribute__((ext_vector_type(16))) _Float16 v16h_t;
typedef __attribute__((ext_vector_type(8)))  _Float16 v8h_t;
typedef __attribute__((ext_vector_type(4)))  _Float16 v4h_t;
typedef __attribute__((ext_vector_type(8)))  float    v8f_t;

#define BB 16
#define HH 128
#define WW 256
#define CC 64
#define KK 5

// One workgroup per batch b (W-recurrence serializes; only B chains are
// independent). 256 threads = 8 wave32 waves. Wave = (mh, nt): fixed N-tile
// nt = wid&3, M half mh = wid>>2 (4 M-tiles each). Each wave keeps its 10
// WMMA B fragments (weights) resident in VGPRs for the whole W loop.
// Column GEMM: M=128, N=64, K_tot=5*64=320 via v_wmma_f32_16x16x32_f16.
// x columns stream HBM->LDS with global_load_async_to_lds_b128 (double
// buffered); g columns stream LDS->HBM with global_store_async_from_lds_b128.
__global__ __launch_bounds__(256) void mv_cnn_left_kernel(
    const float* __restrict__ x, const float* __restrict__ w,
    const float* __restrict__ bias, float* __restrict__ out)
{
    // Weights pre-swizzled into per-lane WMMA B-fragment order:
    // [kk(10)][nt(4)][lane(32)][elem(16)] halfs.
    __shared__ __align__(16) _Float16 wlds[10 * 4 * 32 * 16];   // 40 KB
    // Staged s = x_col + g_prev, f16, 2-row halo top/bottom (zeroed once).
    __shared__ __align__(16) _Float16 slds[(HH + 4) * CC];      // 16.5 KB
    // Recurrent state g (f32) — also the async-store source for out columns.
    __shared__ __align__(16) float    glds[HH * CC];            // 32 KB
    // Double-buffered async staging for x columns (f32).
    __shared__ __align__(16) float    xstage[2 * HH * CC];      // 64 KB

    const int tid  = threadIdx.x;
    const int wid  = tid >> 5;
    const int lane = tid & 31;
    const int hl   = lane >> 4;      // lane half (WMMA fragment layout)
    const int ln15 = lane & 15;
    const int b    = blockIdx.x;
    const int nt   = wid & 3;        // this wave's N-tile (fixed)
    const int mh   = wid >> 2;       // this wave's M half (4 M-tiles)

    const uint32_t glds_lo   = (uint32_t)(uintptr_t)&glds[0];
    const uint32_t xstage_lo = (uint32_t)(uintptr_t)&xstage[0];

    // g_W = 0
    for (int i2 = tid; i2 < HH * CC; i2 += 256) glds[i2] = 0.f;
    // zero halo rows (rows 0,1 and 130,131 of slds) -- never written again
    for (int i2 = tid; i2 < 2 * CC; i2 += 256) {
        slds[i2] = (_Float16)0.f;
        slds[(HH + 2) * CC + i2] = (_Float16)0.f;
    }
    // Stage weights f32->f16 into B-fragment layout.
    // B[K_tot, n] = w[ci][k][0][n], flat w index = ci*(K*C) + k*C + n.
    // Per lane, element e covers local-k lk = (e<8 ? e : e+8) + 8*half.
    for (int idx = tid; idx < KK * CC * CC; idx += 256) {
        int kk   = idx >> 11;
        int rem  = idx & 2047;
        int ntt  = rem >> 9;
        int le   = rem & 511;
        int lnn  = le >> 4;
        int e    = le & 15;
        int hsel = lnn >> 4;
        int n    = ntt * 16 + (lnn & 15);
        int lk   = ((e < 8) ? e : (e + 8)) + 8 * hsel;   // 0..31
        int k    = kk >> 1;
        int ci   = ((kk & 1) << 5) + lk;
        wlds[idx] = (_Float16)w[ci * (KK * CC) + k * CC + n];
    }

    const float* xb = x   + (size_t)b * HH * WW * CC;
    float*       ob = out + (size_t)b * HH * WW * CC;
    const float  bias_v = bias[nt * 16 + ln15];

    // Prologue: async-stage column W-1 into xstage buffer 0.
    {
        const int j = WW - 1;
        for (int f = tid; f < (HH * CC) / 4; f += 256) {
            int h  = f >> 4;
            int c4 = (f & 15) << 2;
            const float* gsrc = xb + ((size_t)h * WW + j) * CC + c4;
            uint32_t     ldst = xstage_lo + (uint32_t)(h * CC + c4) * 4u;
            asm volatile("global_load_async_to_lds_b128 %0, %1, off"
                         :: "v"(ldst), "v"((uint64_t)(uintptr_t)gsrc)
                         : "memory");
        }
    }

    __syncthreads();   // wlds staged

    // Load this wave's 10 B fragments once; resident for the whole W loop.
    v16h_t bfr[10];
    #pragma unroll
    for (int kk = 0; kk < 10; ++kk)
        bfr[kk] = *(const v16h_t*)(wlds + ((kk * 4 + nt) * 32 + lane) * 16);

    int cur = 0;
    for (int i = WW - 1; i >= 0; --i) {
        // Wait for this wave's async ops (loads of column i into xstage[cur],
        // stores of column i+1 out of glds); barrier makes them global.
        asm volatile("s_wait_asynccnt 0x0" ::: "memory");
        __syncthreads();

        // ---- stage s[h+2][c] = f16(xstage[cur][h][c] + g[h][c]) ----
        const float* xcur = xstage + cur * HH * CC;
        for (int f = tid; f < (HH * CC) / 4; f += 256) {
            int h  = f >> 4;
            int c4 = (f & 15) << 2;
            const float4 xv = *(const float4*)(xcur + h * CC + c4);
            const float4 gv = *(const float4*)(glds + h * CC + c4);
            v4h_t sv;
            sv[0] = (_Float16)(xv.x + gv.x);
            sv[1] = (_Float16)(xv.y + gv.y);
            sv[2] = (_Float16)(xv.z + gv.z);
            sv[3] = (_Float16)(xv.w + gv.w);
            *(v4h_t*)(slds + (h + 2) * CC + c4) = sv;
        }
        // Async-stage column i-1 into the other buffer while i computes.
        if (i > 0) {
            const int nb = cur ^ 1;
            for (int f = tid; f < (HH * CC) / 4; f += 256) {
                int h  = f >> 4;
                int c4 = (f & 15) << 2;
                const float* gsrc = xb + ((size_t)h * WW + (i - 1)) * CC + c4;
                uint32_t     ldst = xstage_lo +
                                    (uint32_t)(nb * HH * CC + h * CC + c4) * 4u;
                asm volatile("global_load_async_to_lds_b128 %0, %1, off"
                             :: "v"(ldst), "v"((uint64_t)(uintptr_t)gsrc)
                             : "memory");
            }
        }
        __syncthreads();   // slds ready; stage reads of glds done

        // ---- column GEMM: 4 M-tiles for this wave's fixed nt ----
        #pragma unroll
        for (int j = 0; j < 4; ++j) {
            const int mt = mh * 4 + j;
            v8f_t acc = {};
            #pragma unroll
            for (int kk = 0; kk < 10; ++kk) {
                int k   = kk >> 1;
                int cib = (kk & 1) << 5;
                // A[m, k*64+ci] = s_col[m + k - 2][ci] = slds[m + k][ci]
                const _Float16* arow =
                    slds + (mt * 16 + ln15 + k) * CC + cib + 8 * hl;
                v8h_t alo = *(const v8h_t*)(arow);        // lk 0..7  (+8*hl)
                v8h_t ahi = *(const v8h_t*)(arow + 16);   // lk 16..23(+8*hl)
                v16h_t a  = __builtin_shufflevector(alo, ahi,
                              0,1,2,3,4,5,6,7,8,9,10,11,12,13,14,15);
                acc = __builtin_amdgcn_wmma_f32_16x16x32_f16(
                          false, a, false, bfr[kk], (short)0, acc,
                          false, false);
            }
            // epilogue: g = relu(acc + b) -> glds only (HBM via async store)
            #pragma unroll
            for (int v = 0; v < 8; ++v) {
                int   m = mt * 16 + v + 8 * hl;   // D layout: M = v + 8*half
                int   n = nt * 16 + ln15;
                float e = acc[v] + bias_v;
                glds[m * CC + n] = e > 0.f ? e : 0.f;
            }
        }
        __syncthreads();   // glds (g_i) complete before async store reads it

        // ---- async store column i: out[:, i, :] <= glds ----
        for (int f = tid; f < (HH * CC) / 4; f += 256) {
            int h  = f >> 4;
            int c4 = (f & 15) << 2;
            float*   gdst = ob + ((size_t)h * WW + i) * CC + c4;
            uint32_t lsrc = glds_lo + (uint32_t)(h * CC + c4) * 4u;
            asm volatile("global_store_async_from_lds_b128 %0, %1, off"
                         :: "v"((uint64_t)(uintptr_t)gdst), "v"(lsrc)
                         : "memory");
        }
        cur ^= 1;
    }
    // S_ENDPGM performs an implicit wait-idle (all counters, incl. ASYNCcnt).
}

extern "C" void kernel_launch(void* const* d_in, const int* in_sizes, int n_in,
                              void* d_out, int out_size, void* d_ws, size_t ws_size,
                              hipStream_t stream) {
    const float* x    = (const float*)d_in[0];  // inputs [16,128,256,64] f32
    const float* w    = (const float*)d_in[1];  // w [64,5,1,64] f32
    const float* bias = (const float*)d_in[2];  // b [64] f32
    float*       out  = (float*)d_out;          // [16,128,256,64] f32
    mv_cnn_left_kernel<<<BB, 256, 0, stream>>>(x, w, bias, out);
}
